// QuantizerEMA_23244363006433
// MI455X (gfx1250) — matrix operand order repaired and checked by reading, not compiled, not run on hardware
//
#include <hip/hip_runtime.h>
#include <hip/hip_bf16.h>

// VQ-VAE EMA quantizer for gfx1250 (MI455X).
// Distance matmul on v_wmma_f32_16x16x32_bf16; each wave owns a 32x512
// distance tile (two 16-row A tiles sharing every B fragment from LDS).
// EMA statistics stay exact fp32.

typedef __attribute__((ext_vector_type(16))) __bf16 v16bf;
typedef __attribute__((ext_vector_type(8)))  float  v8f;

#define EMB   64
#define NCODE 512
#define TPTS  (128 * 32 * 32)   // 131072 spatial points
#define HW    1024              // 32*32
#define CHW   (EMB * HW)        // 65536
#define DECAYF 0.99f
#define OMDF   0.01f
#define EPSF   1.0e8f

// d_out flat layout (floats): z_q | e_new | N_new | m_new
#define OFF_E 8388608           // TPTS*EMB
#define OFF_N 8421376           // OFF_E + NCODE*EMB
#define OFF_M 8421888           // OFF_N + NCODE

// workspace byte offsets
#define WS_IDX   0              // int[TPTS]              = 524288 B
#define WS_NACC  524288         // float[NCODE]           =   2048 B
#define WS_MACC  526336         // float[NCODE*EMB]       = 131072 B
#define WS_ESQ   657408         // float[NCODE]           =   2048 B
#define WS_EBF   659456         // ushort[NCODE*EMB]      =  65536 B

static __device__ __forceinline__ unsigned short f32_to_bf16_bits(float f) {
  union { float f; unsigned u; } v; v.f = f;
  unsigned u = v.u;
  unsigned r = u + 0x7FFFu + ((u >> 16) & 1u);   // round-to-nearest-even
  if ((u & 0x7F800000u) == 0x7F800000u) r = u;   // Inf/NaN passthrough
  return (unsigned short)(r >> 16);
}

// ---------------------------------------------------------------------------
// Kernel 0: zero accumulators, build bf16 codebook copy + ||e_k||^2
// ---------------------------------------------------------------------------
__global__ void vq_init(const float* __restrict__ e,
                        float* __restrict__ n_acc,
                        float* __restrict__ m_acc,
                        float* __restrict__ esq,
                        unsigned short* __restrict__ ebf) {
  int tid = blockIdx.x * blockDim.x + threadIdx.x;
  if (tid < NCODE * EMB) {
    m_acc[tid] = 0.0f;
    ebf[tid]   = f32_to_bf16_bits(e[tid]);
  }
  if (tid < NCODE) {
    n_acc[tid] = 0.0f;
    float s = 0.0f;
#pragma unroll 8
    for (int c = 0; c < EMB; ++c) { float v = e[tid * EMB + c]; s += v * v; }
    esq[tid] = s;
  }
}

// ---------------------------------------------------------------------------
// Kernel 1: WMMA distance + argmin + segment sums.
// Wave = two 16-row tiles of z_flat (32 rows); workgroup (8 waves) = 256 rows.
// Entire bf16 codebook (64KB) staged into LDS once per workgroup; every
// B fragment read feeds 2 WMMAs (one per A tile).
// ---------------------------------------------------------------------------
__global__ void __launch_bounds__(256)
vq_assign(const float* __restrict__ z,            // [B, C, H, W] fp32
          const unsigned short* __restrict__ ebf, // [NCODE, EMB] bf16 bits
          const float* __restrict__ esq,          // [NCODE] ||e||^2
          int* __restrict__ idx,                  // [TPTS]
          float* __restrict__ n_acc,              // [NCODE]
          float* __restrict__ m_acc) {            // [NCODE, EMB]
  __shared__ __align__(16) unsigned short sE[NCODE * EMB];  // 64 KB
  {
    const uint4* g = reinterpret_cast<const uint4*>(ebf);
    uint4*       s = reinterpret_cast<uint4*>(sE);
    for (int i = threadIdx.x; i < (NCODE * EMB * 2) / 16; i += blockDim.x)
      s[i] = g[i];
  }
  __syncthreads();

  const int lane  = threadIdx.x & 31;
  const int wave  = threadIdx.x >> 5;
  const int row16 = lane & 15;          // A-row / B-column slot / D-column
  const int half  = (lane >> 4) & 1;    // lane half selects K / M sub-block
  const int t0    = (blockIdx.x * 8 + wave) * 32;   // 32 rows per wave

  // --- A fragments: two row tiles, K = 0..63 as 16x32 bf16 pieces ----------
  // A-layout: lane l holds row M=l%16; element j -> K = half*16 + j
  // (contiguous, matching B).
  union AV { v16bf v; unsigned short s[16]; } a0, a1, a2, a3;
  {
    const int   tA   = t0 + row16;
    const int   tB   = t0 + 16 + row16;
    const float* zA = z + (size_t)(tA >> 10) * CHW + (tA & (HW - 1));
    const float* zB = z + (size_t)(tB >> 10) * CHW + (tB & (HW - 1));
#pragma unroll
    for (int j = 0; j < 16; ++j) {
      int k = (half << 4) + j;
      a0.s[j] = f32_to_bf16_bits(zA[(size_t)k * HW]);
      a1.s[j] = f32_to_bf16_bits(zA[(size_t)(k + 32) * HW]);
      a2.s[j] = f32_to_bf16_bits(zB[(size_t)k * HW]);
      a3.s[j] = f32_to_bf16_bits(zB[(size_t)(k + 32) * HW]);
    }
  }

  float best0[8], best1[8];
  int   bidx0[8], bidx1[8];
#pragma unroll
  for (int r = 0; r < 8; ++r) {
    best0[r] = 3.4e38f; bidx0[r] = 0;
    best1[r] = 3.4e38f; bidx1[r] = 0;
  }

  // --- sweep 32 column tiles of 16 codes: 4 WMMAs per tile -----------------
#pragma unroll 2
  for (int nt = 0; nt < NCODE / 16; ++nt) {
    const int code = nt * 16 + row16;   // this lane's N column (B layout)
    // B fragments as loaded (KxN, 32x16 bf16; contiguous k, same as A): lane n%16 = column; lanes<16 K 0..15,
    // lanes>=16 K 16..31 -> 16 contiguous bf16 per lane per fragment.
    union BV { v16bf v; uint4 q[2]; } b0, b1;
    const uint4* p0 = reinterpret_cast<const uint4*>(&sE[code * EMB + half * 16]);
    b0.q[0] = p0[0]; b0.q[1] = p0[1];
    const uint4* p1 = reinterpret_cast<const uint4*>(&sE[code * EMB + 32 + half * 16]);
    b1.q[0] = p1[0]; b1.q[1] = p1[1];

    const float ee = esq[code];         // ||e_code||^2 (f32, exact)

    v8f acc0 = {}, acc1 = {};
    // consecutive WMMAs share the B operand (operand-reuse friendly)
    acc0 = __builtin_amdgcn_wmma_f32_16x16x32_bf16(false, a0.v, false, b0.v,
                                                   (short)0, acc0, false, false);
    acc1 = __builtin_amdgcn_wmma_f32_16x16x32_bf16(false, a2.v, false, b0.v,
                                                   (short)0, acc1, false, false);
    acc0 = __builtin_amdgcn_wmma_f32_16x16x32_bf16(false, a1.v, false, b1.v,
                                                   (short)0, acc0, false, false);
    acc1 = __builtin_amdgcn_wmma_f32_16x16x32_bf16(false, a3.v, false, b1.v,
                                                   (short)0, acc1, false, false);

#pragma unroll
    for (int r = 0; r < 8; ++r) {       // D layout: VGPR r -> M = r + 8*half
      float d0 = ee - 2.0f * acc0[r];   // ||z||^2 omitted (row-constant)
      float d1 = ee - 2.0f * acc1[r];
      if (d0 < best0[r]) { best0[r] = d0; bidx0[r] = code; }
      if (d1 < best1[r]) { best1[r] = d1; bidx1[r] = code; }
    }
  }

  // --- reduce argmin across the 16 column-lanes of each half ---------------
#pragma unroll
  for (int r = 0; r < 8; ++r) {
#pragma unroll
    for (int off = 1; off < 16; off <<= 1) {
      float ov0 = __shfl_xor(best0[r], off, 32);
      int   oi0 = __shfl_xor(bidx0[r], off, 32);
      if (ov0 < best0[r] || (ov0 == best0[r] && oi0 < bidx0[r])) {
        best0[r] = ov0; bidx0[r] = oi0;
      }
      float ov1 = __shfl_xor(best1[r], off, 32);
      int   oi1 = __shfl_xor(bidx1[r], off, 32);
      if (ov1 < best1[r] || (ov1 == best1[r] && oi1 < bidx1[r])) {
        best1[r] = ov1; bidx1[r] = oi1;
      }
    }
  }

  // lane 0 owns rows 0..7 of each tile, lane 16 owns rows 8..15
  if (row16 == 0) {
#pragma unroll
    for (int r = 0; r < 8; ++r) {
      int m = r + 8 * half;
      idx[t0 + m]      = bidx0[r];
      idx[t0 + 16 + m] = bidx1[r];
      atomicAdd(&n_acc[bidx0[r]], 1.0f);
      atomicAdd(&n_acc[bidx1[r]], 1.0f);
    }
  }

  // --- m_i scatter: all 32 lanes, 2 channels per lane per row --------------
#pragma unroll
  for (int mrow = 0; mrow < 32; ++mrow) {
    int r   = mrow & 7;
    int src = ((mrow & 15) < 8) ? 0 : 16;
    int k   = (mrow < 16) ? __shfl(bidx0[r], src, 32)
                          : __shfl(bidx1[r], src, 32);
    int tt = t0 + mrow;
    int tb = tt >> 10;
    int th = tt & (HW - 1);
    const float* zr = z + (size_t)tb * CHW + th;
    int c0 = lane * 2;
    atomicAdd(&m_acc[k * EMB + c0],     zr[(size_t)c0 * HW]);
    atomicAdd(&m_acc[k * EMB + c0 + 1], zr[(size_t)(c0 + 1) * HW]);
  }
}

// ---------------------------------------------------------------------------
// Kernel 2: exact fp32 EMA update; writes e_new / N_new / m_new slices
// ---------------------------------------------------------------------------
__global__ void vq_update(const float* __restrict__ Nold,
                          const float* __restrict__ mold,
                          const float* __restrict__ n_acc,
                          const float* __restrict__ m_acc,
                          float* __restrict__ out) {
  int tid = blockIdx.x * blockDim.x + threadIdx.x;
  if (tid >= NCODE * EMB) return;
  int k = tid >> 6;
  float Nn = DECAYF * Nold[k] + OMDF * n_acc[k];
  float mn = DECAYF * mold[tid] + OMDF * m_acc[tid];
  float en = mn / (Nn + EPSF);
  out[OFF_E + tid] = en;
  out[OFF_M + tid] = mn;
  if ((tid & 63) == 0) out[OFF_N + k] = Nn;
}

// ---------------------------------------------------------------------------
// Kernel 3: z_q = e_new[idx], NCHW layout; t-fastest for coalesced writes
// (e_new is 128KB -> fully L2-resident, scattered reads are cheap)
// ---------------------------------------------------------------------------
__global__ void vq_gather(const int* __restrict__ idx,
                          const float* __restrict__ enew_base, // d_out base
                          float* __restrict__ zq) {
  int tid = blockIdx.x * blockDim.x + threadIdx.x;  // over EMB*TPTS
  int c = tid >> 17;                                // TPTS = 2^17
  int t = tid & (TPTS - 1);
  int k = idx[t];
  float v = enew_base[OFF_E + k * EMB + c];
  int b  = t >> 10;
  int hw = t & (HW - 1);
  zq[(size_t)b * CHW + (size_t)c * HW + hw] = v;
}

// ---------------------------------------------------------------------------
extern "C" void kernel_launch(void* const* d_in, const int* in_sizes, int n_in,
                              void* d_out, int out_size, void* d_ws, size_t ws_size,
                              hipStream_t stream) {
  const float* z    = (const float*)d_in[0];   // [128,64,32,32]
  const float* e    = (const float*)d_in[1];   // [512,64]
  const float* Nold = (const float*)d_in[2];   // [512]
  const float* mold = (const float*)d_in[3];   // [512,64]
  float* out = (float*)d_out;

  char* ws = (char*)d_ws;
  int*            idx   = (int*)(ws + WS_IDX);
  float*          n_acc = (float*)(ws + WS_NACC);
  float*          m_acc = (float*)(ws + WS_MACC);
  float*          esq   = (float*)(ws + WS_ESQ);
  unsigned short* ebf   = (unsigned short*)(ws + WS_EBF);

  vq_init<<<(NCODE * EMB + 255) / 256, 256, 0, stream>>>(e, n_acc, m_acc, esq, ebf);
  vq_assign<<<TPTS / 256, 256, 0, stream>>>(z, ebf, esq, idx, n_acc, m_acc);
  vq_update<<<(NCODE * EMB + 255) / 256, 256, 0, stream>>>(Nold, mold, n_acc, m_acc, out);
  vq_gather<<<(TPTS * EMB) / 256, 256, 0, stream>>>(idx, out, out);
}
